// BaseX2HAttLayer_66391604461749
// MI455X (gfx1250) — compile-verified, matrix operand order checked by reading, not run on hardware
//
#include <hip/hip_runtime.h>
#include <hip/hip_bf16.h>

typedef __attribute__((ext_vector_type(16))) __bf16 v16bf;
typedef __attribute__((ext_vector_type(8)))  float  v8f;

// ---------- scalar helpers ----------
__device__ __forceinline__ unsigned short f2bf_bits(float x) {
  union { float f; unsigned u; } v; v.f = x;
  unsigned r = v.u + 0x7FFFu + ((v.u >> 16) & 1u);   // round-to-nearest-even
  return (unsigned short)(r >> 16);
}
__device__ __forceinline__ float bf2f(unsigned short b) {
  union { unsigned u; float f; } v; v.u = ((unsigned)b) << 16;
  return v.f;
}
__device__ __forceinline__ v8f v8f_zero() {
  v8f z;
  #pragma unroll
  for (int i = 0; i < 8; ++i) z[i] = 0.0f;
  return z;
}

union Frag16 { uint4 q[2]; v16bf v; };

// A fragment (16x32 bf16) from LDS, row-major with given stride (halfs).
// ISA layout: lanes 0-15: M=lane, K=k0..k0+7 and k0+16..k0+23;
//             lanes 16-31: M=lane-16, K=k0+8..k0+15 and k0+24..k0+31.
__device__ __forceinline__ v16bf load_a_frag_lds(const unsigned short* s, int stride,
                                                 int k0, int lane) {
  int row  = lane & 15;
  int koff = k0 + ((lane >> 4) << 3);
  const unsigned short* p = s + row * stride + koff;
  Frag16 f;
  f.q[0] = *(const uint4*)(p);
  f.q[1] = *(const uint4*)(p + 16);
  return f.v;
}

// B fragment (32x16 bf16) from the pre-packed weight buffer:
// packed[nt][kt][lane][0..15], each tile 32 lanes * 32B contiguous.
__device__ __forceinline__ v16bf load_b_frag(const unsigned short* packed, int KT,
                                             int nt, int kt, int lane) {
  const unsigned short* p = packed + ((((size_t)(nt * KT + kt)) * 32 + lane) * 16);
  Frag16 f;
  f.q[0] = ((const uint4*)p)[0];
  f.q[1] = ((const uint4*)p)[1];
  return f.v;
}

__device__ __forceinline__ v8f wmma_bf16(v16bf a, v16bf b, v8f c) {
  return __builtin_amdgcn_wmma_f32_16x16x32_bf16(false, a, false, b, (short)0, c,
                                                 false, false);
}

// Store 16x16 f32 accumulator (+bias) into LDS fp32 tile of row-stride 256.
// D layout: VGPR r, lanes 0-15 -> (M=r, N=lane); lanes 16-31 -> (M=8+r, N=lane-16).
__device__ __forceinline__ void acc_to_lds(float* sF, v8f acc, int nt, int lane,
                                           const float* __restrict__ bias) {
  int mBase = (lane >> 4) << 3;
  int n = (nt << 4) + (lane & 15);
  float bv = bias[n];
  #pragma unroll
  for (int r = 0; r < 8; ++r) sF[(mBase + r) * 256 + n] = acc[r] + bv;
}

// Cross-wave LayerNorm (+ReLU) over 256-wide rows of a 16x256 fp32 LDS tile,
// writing bf16 result into sOut. blockDim.x must be 256. Contains barriers.
__device__ __forceinline__ void layernorm_relu_256(
    float* sF, unsigned short* sOut,
    const float* __restrict__ lng, const float* __restrict__ lnb,
    float* psum, float* psq, float* smean, float* srstd) {
  int t = threadIdx.x;
  int rm = t >> 4, seg = t & 15;
  const float* row = sF + rm * 256 + seg * 16;
  float s = 0.f, sq = 0.f;
  #pragma unroll
  for (int j = 0; j < 16; ++j) { float x = row[j]; s += x; sq += x * x; }
  psum[rm * 16 + seg] = s;
  psq[rm * 16 + seg]  = sq;
  __syncthreads();
  if (seg == 0) {
    float S = 0.f, Q = 0.f;
    #pragma unroll
    for (int j = 0; j < 16; ++j) { S += psum[rm * 16 + j]; Q += psq[rm * 16 + j]; }
    float mu  = S * (1.0f / 256.0f);
    float var = Q * (1.0f / 256.0f) - mu * mu;
    smean[rm] = mu;
    srstd[rm] = rsqrtf(var + 1e-5f);
  }
  __syncthreads();
  for (int i = t; i < 16 * 256; i += 256) {
    int m = i >> 8, c = i & 255;
    float x = (sF[i] - smean[m]) * srstd[m] * lng[c] + lnb[c];
    sOut[i] = f2bf_bits(x > 0.f ? x : 0.f);
  }
  __syncthreads();
}

// ---------- weight pre-pack: fp32 [K,256] -> bf16 WMMA-B fragment order ----------
__global__ void pack_weight_kernel(const float* __restrict__ W, int K, int KT,
                                   unsigned short* __restrict__ dstp) {
  int total = KT * 16 * 32 * 16;
  for (int i = blockIdx.x * blockDim.x + threadIdx.x; i < total;
       i += gridDim.x * blockDim.x) {
    int ii = i;
    int e    = ii & 15; ii >>= 4;
    int lane = ii & 31; ii >>= 5;
    int kt = ii % KT;
    int nt = ii / KT;
    int hi = lane >> 4;
    int n  = nt * 16 + (lane & 15);
    int k  = kt * 32 + (e < 8 ? hi * 8 + e : 16 + hi * 8 + (e - 8));
    float val = (k < K) ? W[(size_t)k * 256 + n] : 0.f;
    dstp[i] = f2bf_bits(val);
  }
}

// ---------- init: smax=-inf, denom=0, agg=0 ----------
__global__ void init_kernel(unsigned int* __restrict__ smax, float* __restrict__ denom,
                            float* __restrict__ agg, int nH, int nAgg) {
  for (int i = blockIdx.x * blockDim.x + threadIdx.x; i < nAgg;
       i += gridDim.x * blockDim.x) {
    agg[i] = 0.f;
    if (i < nH) { smax[i] = 0xFF800000u; denom[i] = 0.f; }
  }
}

// ---------- q = MLP_hq(h), 16 nodes per block, 8 waves x 2 N-tiles ----------
__global__ __launch_bounds__(256) void hq_kernel(
    const float* __restrict__ h,
    const unsigned short* __restrict__ w1p, const float* __restrict__ b1,
    const float* __restrict__ lng, const float* __restrict__ lnb,
    const unsigned short* __restrict__ w2p, const float* __restrict__ b2,
    float* __restrict__ qout) {
  __shared__ __align__(16) unsigned short sA[16 * 256];
  __shared__ __align__(16) float sF[16 * 256];
  __shared__ float psum[256], psq[256], smean[16], srstd[16];
  int t = threadIdx.x, lane = t & 31, wid = t >> 5;
  int node0 = blockIdx.x * 16;

  for (int i = t; i < 16 * 256; i += 256) {
    int m = i >> 8, c = i & 255;
    sA[i] = f2bf_bits(h[(size_t)(node0 + m) * 256 + c]);
  }
  __syncthreads();

  int nt0 = wid * 2, nt1 = nt0 + 1;
  v8f a0 = v8f_zero(), a1 = v8f_zero();
  for (int kt = 0; kt < 8; ++kt) {
    v16bf a = load_a_frag_lds(sA, 256, kt * 32, lane);
    a0 = wmma_bf16(a, load_b_frag(w1p, 8, nt0, kt, lane), a0);
    a1 = wmma_bf16(a, load_b_frag(w1p, 8, nt1, kt, lane), a1);
  }
  __syncthreads();
  acc_to_lds(sF, a0, nt0, lane, b1);
  acc_to_lds(sF, a1, nt1, lane, b1);
  __syncthreads();
  layernorm_relu_256(sF, sA, lng, lnb, psum, psq, smean, srstd);

  v8f c0 = v8f_zero(), c1 = v8f_zero();
  for (int kt = 0; kt < 8; ++kt) {
    v16bf a = load_a_frag_lds(sA, 256, kt * 32, lane);
    c0 = wmma_bf16(a, load_b_frag(w2p, 8, nt0, kt, lane), c0);
    c1 = wmma_bf16(a, load_b_frag(w2p, 8, nt1, kt, lane), c1);
  }
  int mBase = (lane >> 4) << 3;
  {
    int n = nt0 * 16 + (lane & 15); float bv = b2[n];
    #pragma unroll
    for (int r = 0; r < 8; ++r)
      qout[(size_t)(node0 + mBase + r) * 256 + n] = c0[r] + bv;
  }
  {
    int n = nt1 * 16 + (lane & 15); float bv = b2[n];
    #pragma unroll
    for (int r = 0; r < 8; ++r)
      qout[(size_t)(node0 + mBase + r) * 256 + n] = c1[r] + bv;
  }
}

// ---------- fused per-edge hk + hv MLPs + attention scores ----------
__global__ __launch_bounds__(256) void edge_kv_kernel(
    const float* __restrict__ h, const float* __restrict__ r_feat,
    const float* __restrict__ e_feat, const int* __restrict__ src,
    const int* __restrict__ dst, const float* __restrict__ shape,
    const float* __restrict__ topo, const float* __restrict__ e_w,
    const float* __restrict__ qglob,
    const unsigned short* __restrict__ wk1, const float* __restrict__ bk1,
    const float* __restrict__ gk, const float* __restrict__ bkn,
    const unsigned short* __restrict__ wk2, const float* __restrict__ bk2,
    const unsigned short* __restrict__ wv1, const float* __restrict__ bv1,
    const float* __restrict__ gv, const float* __restrict__ bvn,
    const unsigned short* __restrict__ wv2, const float* __restrict__ bv2,
    float* __restrict__ scores, unsigned int* __restrict__ smax,
    unsigned short* __restrict__ vout) {
  __shared__ __align__(16) unsigned short sKV[16 * 992];   // kv_input bf16, K padded to 992
  __shared__ __align__(16) float sF[16 * 256];
  __shared__ __align__(16) unsigned short sA2[16 * 256];
  __shared__ float psum[256], psq[256], smean[16], srstd[16];
  __shared__ int sDst[16], sSrc[16];
  __shared__ float sEw[16];
  int t = threadIdx.x, lane = t & 31, wid = t >> 5;
  int e0 = blockIdx.x * 16;

  if (t < 16) { sSrc[t] = src[e0 + t]; sDst[t] = dst[e0 + t]; sEw[t] = e_w[e0 + t]; }
  __syncthreads();

  // Gather kv_input = [edge_feat | r_feat | h[dst] | h[src] | topo[dst] | shape[dst] | pad]
  for (int i = t; i < 16 * 992; i += 256) {
    int m = i / 992, c = i - m * 992;
    int d = sDst[m];
    float val;
    if (c < 16)       val = e_feat[(size_t)(e0 + m) * 16 + c];
    else if (c < 80)  val = r_feat[(size_t)(e0 + m) * 64 + (c - 16)];
    else if (c < 336) val = h[(size_t)d * 256 + (c - 80)];
    else if (c < 592) val = h[(size_t)sSrc[m] * 256 + (c - 336)];
    else if (c < 848) val = topo[(size_t)d * 256 + (c - 592)];
    else if (c < 976) val = shape[(size_t)d * 128 + (c - 848)];
    else              val = 0.f;
    sKV[i] = f2bf_bits(val);
  }
  __syncthreads();

  int nt0 = wid * 2, nt1 = nt0 + 1;
  // Layer-1 for both hk and hv in one K sweep (A fragment reused 4x).
  v8f ak0 = v8f_zero(), ak1 = v8f_zero(), av0 = v8f_zero(), av1 = v8f_zero();
  for (int kt = 0; kt < 31; ++kt) {
    v16bf a = load_a_frag_lds(sKV, 992, kt * 32, lane);
    ak0 = wmma_bf16(a, load_b_frag(wk1, 31, nt0, kt, lane), ak0);
    ak1 = wmma_bf16(a, load_b_frag(wk1, 31, nt1, kt, lane), ak1);
    av0 = wmma_bf16(a, load_b_frag(wv1, 31, nt0, kt, lane), av0);
    av1 = wmma_bf16(a, load_b_frag(wv1, 31, nt1, kt, lane), av1);
  }
  __syncthreads();

  // ---- K branch ----
  acc_to_lds(sF, ak0, nt0, lane, bk1);
  acc_to_lds(sF, ak1, nt1, lane, bk1);
  __syncthreads();
  layernorm_relu_256(sF, sA2, gk, bkn, psum, psq, smean, srstd);
  v8f ck0 = v8f_zero(), ck1 = v8f_zero();
  for (int kt = 0; kt < 8; ++kt) {
    v16bf a = load_a_frag_lds(sA2, 256, kt * 32, lane);
    ck0 = wmma_bf16(a, load_b_frag(wk2, 8, nt0, kt, lane), ck0);
    ck1 = wmma_bf16(a, load_b_frag(wk2, 8, nt1, kt, lane), ck1);
  }
  __syncthreads();
  acc_to_lds(sF, ck0, nt0, lane, bk2);
  acc_to_lds(sF, ck1, nt1, lane, bk2);
  __syncthreads();

  // scores: one (edge, head) per thread; k stays in LDS, q gathered from ws.
  {
    int m = t >> 4, hd = t & 15;
    int dn = sDst[m];
    const float* qrow = qglob + (size_t)dn * 256 + hd * 16;
    const float* krow = sF + m * 256 + hd * 16;
    float sc = 0.f;
    #pragma unroll
    for (int j = 0; j < 16; ++j) sc += qrow[j] * krow[j];
    sc *= 0.25f;                                  // 1/sqrt(D_HEAD=16)
    scores[(size_t)(e0 + m) * 16 + hd] = sc;
    unsigned int* addr = smax + (size_t)dn * 16 + hd;
    union { float f; unsigned u; int i; } cvt; cvt.f = sc;
    if (sc >= 0.f) atomicMax((int*)addr, cvt.i);  // float-max via ordered int trick
    else           atomicMin(addr, cvt.u);
  }
  __syncthreads();

  // ---- V branch ----
  acc_to_lds(sF, av0, nt0, lane, bv1);
  acc_to_lds(sF, av1, nt1, lane, bv1);
  __syncthreads();
  layernorm_relu_256(sF, sA2, gv, bvn, psum, psq, smean, srstd);
  v8f cv0 = v8f_zero(), cv1 = v8f_zero();
  for (int kt = 0; kt < 8; ++kt) {
    v16bf a = load_a_frag_lds(sA2, 256, kt * 32, lane);
    cv0 = wmma_bf16(a, load_b_frag(wv2, 8, nt0, kt, lane), cv0);
    cv1 = wmma_bf16(a, load_b_frag(wv2, 8, nt1, kt, lane), cv1);
  }
  int mBase = (lane >> 4) << 3;
  {
    int n = nt0 * 16 + (lane & 15); float bv = bv2[n];
    #pragma unroll
    for (int r = 0; r < 8; ++r) {
      int m = mBase + r;
      vout[(size_t)(e0 + m) * 256 + n] = f2bf_bits((cv0[r] + bv) * sEw[m]);
    }
  }
  {
    int n = nt1 * 16 + (lane & 15); float bv = bv2[n];
    #pragma unroll
    for (int r = 0; r < 8; ++r) {
      int m = mBase + r;
      vout[(size_t)(e0 + m) * 256 + n] = f2bf_bits((cv1[r] + bv) * sEw[m]);
    }
  }
}

// ---------- scatter softmax numerators + unnormalized aggregation ----------
__global__ __launch_bounds__(256) void softmax_agg_kernel(
    const float* __restrict__ scores, const float* __restrict__ smaxf,
    const int* __restrict__ dst, const unsigned short* __restrict__ vbf,
    float* __restrict__ denom, float* __restrict__ agg, int E) {
  int t = blockIdx.x * 256 + threadIdx.x;
  if (t >= E * 16) return;
  int e = t >> 4, hd = t & 15;
  int d = dst[e];
  float ex = __expf(scores[t] - smaxf[(size_t)d * 16 + hd]);
  atomicAdd(denom + (size_t)d * 16 + hd, ex);
  const unsigned short* vp = vbf + (size_t)e * 256 + hd * 16;
  float* ap = agg + (size_t)d * 256 + hd * 16;
  #pragma unroll
  for (int j = 0; j < 16; ++j) atomicAdd(ap + j, ex * bf2f(vp[j]));
}

// ---------- out = MLP_node([agg/denom, h]) + h ----------
__global__ __launch_bounds__(256) void node_out_kernel(
    const float* __restrict__ h, const float* __restrict__ agg,
    const float* __restrict__ denom,
    const unsigned short* __restrict__ w1p, const float* __restrict__ b1,
    const float* __restrict__ lng, const float* __restrict__ lnb,
    const unsigned short* __restrict__ w2p, const float* __restrict__ b2,
    float* __restrict__ out) {
  __shared__ __align__(16) unsigned short sIn[16 * 512];
  __shared__ __align__(16) float sF[16 * 256];
  __shared__ __align__(16) unsigned short sA2[16 * 256];
  __shared__ float psum[256], psq[256], smean[16], srstd[16];
  int t = threadIdx.x, lane = t & 31, wid = t >> 5;
  int node0 = blockIdx.x * 16;

  for (int i = t; i < 16 * 512; i += 256) {
    int m = i >> 9, c = i & 511;
    int node = node0 + m;
    float val;
    if (c < 256) {
      float dd = denom[(size_t)node * 16 + (c >> 4)];
      val = (dd > 0.f) ? agg[(size_t)node * 256 + c] / dd : 0.f;
    } else {
      val = h[(size_t)node * 256 + (c - 256)];
    }
    sIn[i] = f2bf_bits(val);
  }
  __syncthreads();

  int nt0 = wid * 2, nt1 = nt0 + 1;
  v8f a0 = v8f_zero(), a1 = v8f_zero();
  for (int kt = 0; kt < 16; ++kt) {
    v16bf a = load_a_frag_lds(sIn, 512, kt * 32, lane);
    a0 = wmma_bf16(a, load_b_frag(w1p, 16, nt0, kt, lane), a0);
    a1 = wmma_bf16(a, load_b_frag(w1p, 16, nt1, kt, lane), a1);
  }
  __syncthreads();
  acc_to_lds(sF, a0, nt0, lane, b1);
  acc_to_lds(sF, a1, nt1, lane, b1);
  __syncthreads();
  layernorm_relu_256(sF, sA2, lng, lnb, psum, psq, smean, srstd);

  v8f c0 = v8f_zero(), c1 = v8f_zero();
  for (int kt = 0; kt < 8; ++kt) {
    v16bf a = load_a_frag_lds(sA2, 256, kt * 32, lane);
    c0 = wmma_bf16(a, load_b_frag(w2p, 8, nt0, kt, lane), c0);
    c1 = wmma_bf16(a, load_b_frag(w2p, 8, nt1, kt, lane), c1);
  }
  int mBase = (lane >> 4) << 3;
  {
    int n = nt0 * 16 + (lane & 15); float bv = b2[n];
    #pragma unroll
    for (int r = 0; r < 8; ++r) {
      int node = node0 + mBase + r;
      out[(size_t)node * 256 + n] = c0[r] + bv + h[(size_t)node * 256 + n];
    }
  }
  {
    int n = nt1 * 16 + (lane & 15); float bv = b2[n];
    #pragma unroll
    for (int r = 0; r < 8; ++r) {
      int node = node0 + mBase + r;
      out[(size_t)node * 256 + n] = c1[r] + bv + h[(size_t)node * 256 + n];
    }
  }
}

extern "C" void kernel_launch(void* const* d_in, const int* in_sizes, int n_in,
                              void* d_out, int out_size, void* d_ws, size_t ws_size,
                              hipStream_t stream) {
  const float* h      = (const float*)d_in[0];
  const float* r_feat = (const float*)d_in[1];
  const float* e_feat = (const float*)d_in[2];
  const int*   eidx   = (const int*)d_in[3];
  const float* shape  = (const float*)d_in[4];
  const float* topo   = (const float*)d_in[5];
  const float* e_w    = (const float*)d_in[6];
  // params flattened in dict insertion order: hk, hv, hq, node_output;
  // each group: w1, b1, lng, lnb, w2, b2
  const float* hk_w1 = (const float*)d_in[7];
  const float* hk_b1 = (const float*)d_in[8];
  const float* hk_g  = (const float*)d_in[9];
  const float* hk_bn = (const float*)d_in[10];
  const float* hk_w2 = (const float*)d_in[11];
  const float* hk_b2 = (const float*)d_in[12];
  const float* hv_w1 = (const float*)d_in[13];
  const float* hv_b1 = (const float*)d_in[14];
  const float* hv_g  = (const float*)d_in[15];
  const float* hv_bn = (const float*)d_in[16];
  const float* hv_w2 = (const float*)d_in[17];
  const float* hv_b2 = (const float*)d_in[18];
  const float* hq_w1 = (const float*)d_in[19];
  const float* hq_b1 = (const float*)d_in[20];
  const float* hq_g  = (const float*)d_in[21];
  const float* hq_bn = (const float*)d_in[22];
  const float* hq_w2 = (const float*)d_in[23];
  const float* hq_b2 = (const float*)d_in[24];
  const float* no_w1 = (const float*)d_in[25];
  const float* no_b1 = (const float*)d_in[26];
  const float* no_g  = (const float*)d_in[27];
  const float* no_bn = (const float*)d_in[28];
  const float* no_w2 = (const float*)d_in[29];
  const float* no_b2 = (const float*)d_in[30];

  int E = in_sizes[6];          // e_w has E elements
  int N = in_sizes[0] / 256;    // h is [N,256]
  const int* src  = eidx;
  const int* dstp = eidx + E;

  char* ws = (char*)d_ws;
  size_t off = 0;
  auto alloc = [&](size_t bytes) {
    size_t o = off;
    off += (bytes + 255) & ~(size_t)255;
    return o;
  };
  // packed weights: bytes = KT * 16 nt-tiles * 32 lanes * 16 halfs * 2B = KT*16384
  size_t o_wq1 = alloc((size_t)8  * 16384);
  size_t o_wq2 = alloc((size_t)8  * 16384);
  size_t o_wk1 = alloc((size_t)31 * 16384);
  size_t o_wk2 = alloc((size_t)8  * 16384);
  size_t o_wv1 = alloc((size_t)31 * 16384);
  size_t o_wv2 = alloc((size_t)8  * 16384);
  size_t o_wn1 = alloc((size_t)16 * 16384);
  size_t o_wn2 = alloc((size_t)8  * 16384);
  size_t o_q   = alloc((size_t)N * 256 * 4);   // q fp32
  size_t o_sc  = alloc((size_t)E * 16 * 4);    // scores fp32
  size_t o_mx  = alloc((size_t)N * 16 * 4);    // segment max
  size_t o_dn  = alloc((size_t)N * 16 * 4);    // softmax denominators
  size_t o_ag  = alloc((size_t)N * 256 * 4);   // unnormalized aggregation
  size_t o_v   = alloc((size_t)E * 256 * 2);   // v bf16
  (void)ws_size; (void)n_in; (void)out_size;

  auto launch_pack = [&](const float* W, int K, int KT, size_t o) {
    int total = KT * 8192;
    pack_weight_kernel<<<(total + 255) / 256, 256, 0, stream>>>(
        W, K, KT, (unsigned short*)(ws + o));
  };
  launch_pack(hq_w1, 256, 8,  o_wq1);
  launch_pack(hq_w2, 256, 8,  o_wq2);
  launch_pack(hk_w1, 976, 31, o_wk1);
  launch_pack(hk_w2, 256, 8,  o_wk2);
  launch_pack(hv_w1, 976, 31, o_wv1);
  launch_pack(hv_w2, 256, 8,  o_wv2);
  launch_pack(no_w1, 512, 16, o_wn1);
  launch_pack(no_w2, 256, 8,  o_wn2);

  init_kernel<<<(N * 256 + 255) / 256, 256, 0, stream>>>(
      (unsigned int*)(ws + o_mx), (float*)(ws + o_dn), (float*)(ws + o_ag),
      N * 16, N * 256);

  hq_kernel<<<N / 16, 256, 0, stream>>>(
      h, (const unsigned short*)(ws + o_wq1), hq_b1, hq_g, hq_bn,
      (const unsigned short*)(ws + o_wq2), hq_b2, (float*)(ws + o_q));

  edge_kv_kernel<<<E / 16, 256, 0, stream>>>(
      h, r_feat, e_feat, src, dstp, shape, topo, e_w, (const float*)(ws + o_q),
      (const unsigned short*)(ws + o_wk1), hk_b1, hk_g, hk_bn,
      (const unsigned short*)(ws + o_wk2), hk_b2,
      (const unsigned short*)(ws + o_wv1), hv_b1, hv_g, hv_bn,
      (const unsigned short*)(ws + o_wv2), hv_b2,
      (float*)(ws + o_sc), (unsigned int*)(ws + o_mx),
      (unsigned short*)(ws + o_v));

  softmax_agg_kernel<<<(E * 16 + 255) / 256, 256, 0, stream>>>(
      (const float*)(ws + o_sc), (const float*)(ws + o_mx), dstp,
      (const unsigned short*)(ws + o_v), (float*)(ws + o_dn),
      (float*)(ws + o_ag), E);

  node_out_kernel<<<N / 16, 256, 0, stream>>>(
      h, (const float*)(ws + o_ag), (const float*)(ws + o_dn),
      (const unsigned short*)(ws + o_wn1), no_b1, no_g, no_bn,
      (const unsigned short*)(ws + o_wn2), no_b2, (float*)d_out);
}
